// ConditionedS4DTransition_30305289240681
// MI455X (gfx1250) — matrix operand order repaired and checked by reading, not compiled, not run on hardware
//
#include <hip/hip_runtime.h>
#include <math.h>

// Problem dims
#define B_      2048
#define T_      64
#define DYN_    256
#define CD_     128
#define STAT_   128
#define UD_     64
#define NOBS_   80
#define H_      512
#define HZ_     384
#define SEL_IN_ 449
#define K0P_    480   // selector input K padded to multiple of 32

typedef __bf16 bf16;
typedef __bf16 v8bf  __attribute__((ext_vector_type(8)));
typedef __bf16 v16bf __attribute__((ext_vector_type(16)));
typedef float  v8f   __attribute__((ext_vector_type(8)));

__device__ __forceinline__ bf16 f2bf(float f) {
  union { float f; unsigned u; } x; x.f = f;
  unsigned r = x.u + 0x7FFFu + ((x.u >> 16) & 1u);   // RNE
  union { unsigned short s; bf16 b; } y; y.s = (unsigned short)(r >> 16);
  return y.b;
}

__device__ __forceinline__ v8f wmma_bf16(v16bf a, v16bf b, v8f c) {
  // v_wmma_f32_16x16x32_bf16: (neg_a, A, neg_b, B, c_mod, C, reuse_a, reuse_b)
  return __builtin_amdgcn_wmma_f32_16x16x32_bf16(false, a, false, b, (short)0, c, false, false);
}

// A fragment (16x32 bf16, M x K) from LDS tile with row stride `stride` elems.
// lanes 0-15:  M=lane,   K = {0..7, 16..23}
// lanes 16-31: M=lane-16, K = {8..15, 24..31}
__device__ __forceinline__ v16bf load_a_frag(const bf16* lds, int stride, int row16,
                                             int kb, int lane) {
  const int r   = lane & 15;
  const int klo = (lane >> 4) * 8;
  const bf16* p = lds + (row16 + r) * stride + kb + klo;
  v8bf lo = *(const v8bf*)p;
  v8bf hi = *(const v8bf*)(p + 16);
  v16bf a;
#pragma unroll
  for (int i = 0; i < 8; ++i) { a[i] = lo[i]; a[8 + i] = hi[i]; }
  return a;
}

// B fragment (32x16 bf16, K x N) from global transposed weights WT[N][K].
// lane n in 0-15: col n, K={0..7,16..23}; lane n+16: col n, K={8..15,24..31}
__device__ __forceinline__ v16bf load_b_frag(const bf16* __restrict__ WT, int K,
                                             int nBase, int kb, int lane) {
  const int n   = nBase + (lane & 15);
  const int klo = (lane >> 4) * 8;
  const bf16* p = WT + (size_t)n * K + kb + klo;
  v8bf lo = *(const v8bf*)p;
  v8bf hi = *(const v8bf*)(p + 16);
  v16bf b;
#pragma unroll
  for (int i = 0; i < 8; ++i) { b[i] = lo[i]; b[8 + i] = hi[i]; }
  return b;
}

// ---------------------------------------------------------------------------
// GEMM: out[M][N] = act(A[M][K](bf16) @ WT[N][K](bf16)^T + bias)
// Block tile 128x128, 8 wave32s (4 M-waves x 2 N-waves), wave tile 32x64.
// A staged through LDS; B streamed from L2 (weights resident).
// ---------------------------------------------------------------------------
__global__ __launch_bounds__(256) void gemm_bf16(
    const bf16* __restrict__ A, int lda,
    const bf16* __restrict__ WT, int K,
    const float* __restrict__ bias,
    void* __restrict__ out, int ldo,
    int relu, int out_fp32)
{
  __shared__ bf16 As[128 * 40];                    // 128 rows x 32 (+8 pad)
  const int lane  = threadIdx.x & 31;
  const int wave  = threadIdx.x >> 5;
  const int waveM = wave >> 1, waveN = wave & 1;
  const int mBlock = blockIdx.y * 128;
  const int nBlock = blockIdx.x * 128;

  v8f acc[2][4];
#pragma unroll
  for (int i = 0; i < 2; ++i)
#pragma unroll
    for (int j = 0; j < 4; ++j) acc[i][j] = (v8f){0,0,0,0,0,0,0,0};

  for (int kb = 0; kb < K; kb += 32) {
    __syncthreads();
    {   // stage A tile: each thread 16 elems (32B)
      const int r = threadIdx.x >> 1;
      const int h = (threadIdx.x & 1) * 16;
      const bf16* src = A + (size_t)(mBlock + r) * lda + kb + h;
      v8bf x0 = *(const v8bf*)src;
      v8bf x1 = *(const v8bf*)(src + 8);
      *(v8bf*)&As[r * 40 + h]     = x0;
      *(v8bf*)&As[r * 40 + h + 8] = x1;
    }
    __syncthreads();

    v16bf afr[2];
    afr[0] = load_a_frag(As, 40, waveM * 32,      0, lane);
    afr[1] = load_a_frag(As, 40, waveM * 32 + 16, 0, lane);
#pragma unroll
    for (int ni = 0; ni < 4; ++ni) {
      const int nb = nBlock + waveN * 64 + ni * 16;
      v16bf bfr = load_b_frag(WT, K, nb, kb, lane);
      // prefetch next K-slice of this weight row group into caches
      __builtin_prefetch(WT + (size_t)(nb + (lane & 15)) * K + kb + 32, 0, 1);
      acc[0][ni] = wmma_bf16(afr[0], bfr, acc[0][ni]);
      acc[1][ni] = wmma_bf16(afr[1], bfr, acc[1][ni]);
    }
  }

#pragma unroll
  for (int mi = 0; mi < 2; ++mi)
#pragma unroll
    for (int ni = 0; ni < 4; ++ni) {
      const int n  = nBlock + waveN * 64 + ni * 16 + (lane & 15);
      const float bn = bias ? bias[n] : 0.f;
#pragma unroll
      for (int j = 0; j < 8; ++j) {
        const int m = mBlock + waveM * 32 + mi * 16 + j + ((lane >> 4) << 3);
        float v = acc[mi][ni][j] + bn;
        if (relu) v = fmaxf(v, 0.f);
        if (out_fp32) ((float*)out)[(size_t)m * ldo + n] = v;
        else          ((bf16*)out)[(size_t)m * ldo + n]  = f2bf(v);
      }
    }
}

// ---------------------------------------------------------------------------
// Head contraction on matrix cores:
//   out[b][oc] (+)= sign * sum_c ( hz[b,:] @ WT[oc*NC+c,:] + bias[oc*NC+c] ) * vec[b,c]
// Block: 16 rows x 8 head-channels (one per wave). NC in {64,128}, K = 384.
// ---------------------------------------------------------------------------
__global__ __launch_bounds__(256) void contract_head(
    const bf16* __restrict__ hz,       // [B][384]
    const bf16* __restrict__ WT,       // [(NO*NC)][384]
    const float* __restrict__ bias,    // [NO*NC]
    const float* __restrict__ vec,     // [B][NC]
    float* __restrict__ out, int ldo,
    int NC, float sign, int accumulate)
{
  __shared__ bf16 hzs[16 * 392];       // 16 rows x 384 (+8 pad)
  const int lane = threadIdx.x & 31;
  const int wave = threadIdx.x >> 5;
  const int rowBase = blockIdx.y * 16;

  for (int c = threadIdx.x; c < 16 * 48; c += 256) {
    const int r = c / 48, off = (c % 48) * 8;
    v8bf x = *(const v8bf*)(hz + (size_t)(rowBase + r) * 384 + off);
    *(v8bf*)&hzs[r * 392 + off] = x;
  }
  __syncthreads();

  const int oc = blockIdx.x * 8 + wave;
  const int nT = NC >> 4;              // 4 or 8 (wave-uniform)
  v8f acc[8];
#pragma unroll
  for (int t = 0; t < 8; ++t) acc[t] = (v8f){0,0,0,0,0,0,0,0};

  for (int kb = 0; kb < 384; kb += 32) {
    v16bf a = load_a_frag(hzs, 392, 0, kb, lane);
    for (int t = 0; t < nT; ++t) {
      v16bf b = load_b_frag(WT, 384, oc * NC + t * 16, kb, lane);
      acc[t] = wmma_bf16(a, b, acc[t]);
    }
  }

  // per-lane partial contraction with vec, then butterfly over lanes 0..15
  float P[8] = {0,0,0,0,0,0,0,0};
  const int hi = lane >> 4;
  for (int t = 0; t < nT; ++t) {
    const int c  = t * 16 + (lane & 15);
    const float bb = bias[oc * NC + c];
#pragma unroll
    for (int j = 0; j < 8; ++j) {
      const int row = rowBase + j + (hi << 3);
      P[j] += (acc[t][j] + bb) * vec[(size_t)row * NC + c];
    }
  }
#pragma unroll
  for (int mask = 1; mask < 16; mask <<= 1)
#pragma unroll
    for (int j = 0; j < 8; ++j) P[j] += __shfl_xor(P[j], mask, 32);

  if ((lane & 15) == 0) {
#pragma unroll
    for (int j = 0; j < 8; ++j) {
      const int row = rowBase + j + (hi << 3);
      const float v = sign * P[j];
      float* dst = out + (size_t)row * ldo + oc;
      if (accumulate) *dst += v; else *dst = v;
    }
  }
}

// ---------------------------------------------------------------------------
// Elementwise / prep kernels
// ---------------------------------------------------------------------------
__global__ void build_x(const float* __restrict__ znr, const float* __restrict__ zni,
                        const float* __restrict__ zs,  const float* __restrict__ dtv,
                        const float* __restrict__ Ut,  bf16* __restrict__ X,
                        float* __restrict__ udt)
{
  const int idx = blockIdx.x * blockDim.x + threadIdx.x;
  if (idx >= B_ * K0P_) return;
  const int b = idx / K0P_, col = idx % K0P_;
  float v;
  if      (col < 128) v = znr[b * 128 + col];
  else if (col < 256) v = zni[b * 128 + col - 128];
  else if (col < 384) v = zs[b * 128 + col - 256];
  else if (col < 448) v = Ut[b * 64 + col - 384];
  else if (col == 448) v = dtv[b];
  else v = 0.f;
  X[idx] = f2bf(v);
  if (col < 64) udt[b * 64 + col] = Ut[b * 64 + col] * dtv[b];
}

__global__ void lam_apply(const float* __restrict__ raw,
                          float* __restrict__ znr, float* __restrict__ zni)
{
  const int idx = blockIdx.x * blockDim.x + threadIdx.x;
  if (idx >= B_ * CD_) return;
  const int b = idx / CD_, c = idx % CD_;
  const float a = raw[b * 256 + c];
  const float o = raw[b * 256 + 128 + c];
  const float sp = (a > 20.f) ? a : log1pf(expf(a));   // softplus
  const float r  = expf(-sp);
  const float lr = r * cosf(o), li = r * sinf(o);
  const float zr = znr[idx], zi = zni[idx];
  znr[idx] = lr * zr - li * zi;
  zni[idx] = lr * zi + li * zr;
}

__global__ void emit_z(const float* __restrict__ znr, const float* __restrict__ zni,
                       float* __restrict__ Zt)
{
  const int idx = blockIdx.x * blockDim.x + threadIdx.x;
  if (idx >= B_ * DYN_) return;
  const int b = idx / 256, c = idx % 256;
  Zt[idx] = (c < 128) ? znr[b * 128 + c] : zni[b * 128 + c - 128];
}

__global__ void init_state(const float* __restrict__ z_dyn,
                           float* __restrict__ znr, float* __restrict__ zni)
{
  const int idx = blockIdx.x * blockDim.x + threadIdx.x;
  if (idx >= B_ * 128) return;
  const int b = idx / 128, c = idx % 128;
  znr[idx] = z_dyn[(size_t)b * 256 + c];
  zni[idx] = z_dyn[(size_t)b * 256 + 128 + c];
}

// generic transpose fp32[K][N] -> bf16[N][K]; optional BN scale per output col n
__global__ void transpose_to_bf16(const float* __restrict__ W, int K, int N,
                                  bf16* __restrict__ WT,
                                  const float* __restrict__ g, int scaled)
{
  const int idx = blockIdx.x * blockDim.x + threadIdx.x;
  if (idx >= K * N) return;
  const int n = idx / K, k = idx % K;
  float s = 1.f;
  if (scaled) s = rsqrtf(1.0f + 1e-5f) * g[n];
  WT[idx] = f2bf(W[(size_t)k * N + n] * s);
}

// W0: fp32[449][512] -> bf16[512][480] with BN fold and zero K-padding
__global__ void transpose_pad_w0(const float* __restrict__ W,
                                 const float* __restrict__ g,
                                 bf16* __restrict__ WT)
{
  const int idx = blockIdx.x * blockDim.x + threadIdx.x;
  if (idx >= 512 * 480) return;
  const int n = idx / 480, k = idx % 480;
  const float inv = rsqrtf(1.0f + 1e-5f);
  const float v = (k < SEL_IN_) ? W[(size_t)k * 512 + n] * (inv * g[n]) : 0.f;
  WT[idx] = f2bf(v);
}

__global__ void fold_bias(const float* __restrict__ b, const float* __restrict__ g,
                          const float* __restrict__ be, float* __restrict__ dst, int n)
{
  const int idx = blockIdx.x * blockDim.x + threadIdx.x;
  if (idx >= n) return;
  dst[idx] = b[idx] * rsqrtf(1.0f + 1e-5f) * g[idx] + be[idx];
}

__global__ void concat_bias(const float* __restrict__ ba, const float* __restrict__ bo,
                            float* __restrict__ bao)
{
  const int idx = blockIdx.x * blockDim.x + threadIdx.x;
  if (idx >= 256) return;
  bao[idx] = (idx < 128) ? ba[idx] : bo[idx - 128];
}

// ---------------------------------------------------------------------------
extern "C" void kernel_launch(void* const* d_in, const int* in_sizes, int n_in,
                              void* d_out, int out_size, void* d_ws, size_t ws_size,
                              hipStream_t stream)
{
  (void)in_sizes; (void)n_in; (void)out_size; (void)ws_size;
  const float* z_dyn    = (const float*)d_in[0];
  const float* z_static = (const float*)d_in[1];
  const float* dtv      = (const float*)d_in[2];
  const float* U        = (const float*)d_in[3];
  const float* W0  = (const float*)d_in[4];
  const float* b0  = (const float*)d_in[5];
  const float* g0  = (const float*)d_in[6];
  const float* be0 = (const float*)d_in[7];
  const float* W1  = (const float*)d_in[8];
  const float* b1  = (const float*)d_in[9];
  const float* g1  = (const float*)d_in[10];
  const float* be1 = (const float*)d_in[11];
  const float* W2  = (const float*)d_in[12];
  const float* b2  = (const float*)d_in[13];
  const float* Wa  = (const float*)d_in[14];
  const float* ba  = (const float*)d_in[15];
  const float* Wo  = (const float*)d_in[16];
  const float* bo  = (const float*)d_in[17];
  const float* WBr = (const float*)d_in[18];
  const float* bBr = (const float*)d_in[19];
  const float* WBi = (const float*)d_in[20];
  const float* bBi = (const float*)d_in[21];
  const float* WCr = (const float*)d_in[22];
  const float* bCr = (const float*)d_in[23];
  const float* WCi = (const float*)d_in[24];
  const float* bCi = (const float*)d_in[25];
  const float* WD  = (const float*)d_in[26];
  const float* bD  = (const float*)d_in[27];

  char* ws = (char*)d_ws;
  size_t off = 0;
  auto alloc = [&](size_t bytes) -> void* {
    off = (off + 255) & ~(size_t)255;
    void* p = ws + off;
    off += bytes;
    return p;
  };

  bf16* W0T  = (bf16*)alloc((size_t)512  * 480 * 2);
  bf16* W1T  = (bf16*)alloc((size_t)512  * 512 * 2);
  bf16* W2T  = (bf16*)alloc((size_t)384  * 512 * 2);
  bf16* WaoT = (bf16*)alloc((size_t)256  * 384 * 2);
  bf16* WBrT = (bf16*)alloc((size_t)8192 * 384 * 2);
  bf16* WBiT = (bf16*)alloc((size_t)8192 * 384 * 2);
  bf16* WCrT = (bf16*)alloc((size_t)10240* 384 * 2);
  bf16* WCiT = (bf16*)alloc((size_t)10240* 384 * 2);
  bf16* WDT  = (bf16*)alloc((size_t)5120 * 384 * 2);
  float* b0f = (float*)alloc(512 * 4);
  float* b1f = (float*)alloc(512 * 4);
  float* bao = (float*)alloc(256 * 4);
  bf16* X   = (bf16*)alloc((size_t)B_ * K0P_ * 2);
  bf16* H1  = (bf16*)alloc((size_t)B_ * 512 * 2);
  bf16* H2  = (bf16*)alloc((size_t)B_ * 512 * 2);
  bf16* hz  = (bf16*)alloc((size_t)B_ * 384 * 2);
  float* raw = (float*)alloc((size_t)B_ * 256 * 4);
  float* udt = (float*)alloc((size_t)B_ * 64  * 4);
  float* znr = (float*)alloc((size_t)B_ * 128 * 4);
  float* zni = (float*)alloc((size_t)B_ * 128 * 4);

  const int TB = 256;
  auto g1d = [](long long n) { return dim3((unsigned)((n + 255) / 256)); };

  // ---- one-time prep (re-run every call: deterministic) ----
  transpose_pad_w0<<<g1d(512LL*480), TB, 0, stream>>>(W0, g0, W0T);
  transpose_to_bf16<<<g1d(512LL*512), TB, 0, stream>>>(W1, 512, 512, W1T, g1, 1);
  transpose_to_bf16<<<g1d(512LL*384), TB, 0, stream>>>(W2, 512, 384, W2T, nullptr, 0);
  transpose_to_bf16<<<g1d(384LL*128), TB, 0, stream>>>(Wa, 384, 128, WaoT, nullptr, 0);
  transpose_to_bf16<<<g1d(384LL*128), TB, 0, stream>>>(Wo, 384, 128, WaoT + 128*384, nullptr, 0);
  transpose_to_bf16<<<g1d(384LL*8192), TB, 0, stream>>>(WBr, 384, 8192, WBrT, nullptr, 0);
  transpose_to_bf16<<<g1d(384LL*8192), TB, 0, stream>>>(WBi, 384, 8192, WBiT, nullptr, 0);
  transpose_to_bf16<<<g1d(384LL*10240),TB, 0, stream>>>(WCr, 384, 10240, WCrT, nullptr, 0);
  transpose_to_bf16<<<g1d(384LL*10240),TB, 0, stream>>>(WCi, 384, 10240, WCiT, nullptr, 0);
  transpose_to_bf16<<<g1d(384LL*5120), TB, 0, stream>>>(WD, 384, 5120, WDT, nullptr, 0);
  fold_bias<<<g1d(512), TB, 0, stream>>>(b0, g0, be0, b0f, 512);
  fold_bias<<<g1d(512), TB, 0, stream>>>(b1, g1, be1, b1f, 512);
  concat_bias<<<1, 256, 0, stream>>>(ba, bo, bao);
  init_state<<<g1d((long long)B_*128), TB, 0, stream>>>(z_dyn, znr, zni);

  float* Zout = (float*)d_out;                               // [T][B][256]
  float* Yout = (float*)d_out + (size_t)T_ * B_ * DYN_;      // [T][B][80]

  // ---- sequential scan over T ----
  for (int t = 0; t < T_; ++t) {
    const float* Ut = U + (size_t)t * B_ * UD_;
    build_x<<<g1d((long long)B_*K0P_), TB, 0, stream>>>(znr, zni, z_static, dtv, Ut, X, udt);

    gemm_bf16<<<dim3(4, 16), TB, 0, stream>>>(X,  K0P_, W0T, K0P_, b0f, H1, 512, 1, 0);
    gemm_bf16<<<dim3(4, 16), TB, 0, stream>>>(H1, 512,  W1T, 512,  b1f, H2, 512, 1, 0);
    gemm_bf16<<<dim3(3, 16), TB, 0, stream>>>(H2, 512,  W2T, 512,  b2,  hz, 384, 0, 0);
    gemm_bf16<<<dim3(2, 16), TB, 0, stream>>>(hz, 384,  WaoT, 384, bao, raw, 256, 0, 1);

    lam_apply<<<g1d((long long)B_*CD_), TB, 0, stream>>>(raw, znr, zni);

    // zn += Bt @ (u*dt)   (real / imag)
    contract_head<<<dim3(16, 128), TB, 0, stream>>>(hz, WBrT, bBr, udt, znr, 128, 64, 1.f, 1);
    contract_head<<<dim3(16, 128), TB, 0, stream>>>(hz, WBiT, bBi, udt, zni, 128, 64, 1.f, 1);

    emit_z<<<g1d((long long)B_*DYN_), TB, 0, stream>>>(znr, zni, Zout + (size_t)t * B_ * DYN_);

    // y = Re(Ct @ zn) + Dt @ (u*dt)
    float* Yt = Yout + (size_t)t * B_ * NOBS_;
    contract_head<<<dim3(10, 128), TB, 0, stream>>>(hz, WCrT, bCr, znr, Yt, NOBS_, 128,  1.f, 0);
    contract_head<<<dim3(10, 128), TB, 0, stream>>>(hz, WCiT, bCi, zni, Yt, NOBS_, 128, -1.f, 1);
    contract_head<<<dim3(10, 128), TB, 0, stream>>>(hz, WDT,  bD,  udt, Yt, NOBS_, 64,   1.f, 1);
  }
}
